// Net_kv75_74637941669906
// MI455X (gfx1250) — matrix operand order repaired and checked by reading, not compiled, not run on hardware
//
#include <hip/hip_runtime.h>

// ---------------------------------------------------------------------------
// MI455X (gfx1250) masked dense 3D CNN.
// Implicit-GEMM conv on v_wmma_f32_16x16x32_f16:
//   - wave tile: 16 voxels x 32 output channels (2 accumulators, shared A)
//   - block: 8 M-tiles x one 32-cout N-group
//   - weights staged per-tap into LDS with GLOBAL_LOAD_ASYNC_TO_LDS_B128
//     (ASYNCcnt + s_wait_asynccnt), consumed via ds_load_b128 by all 8 waves
//   - OOB conv taps handled by a zero-page address clamp (no EXEC churn,
//     no per-iteration zero fills in the inner loop)
// ---------------------------------------------------------------------------

typedef _Float16 half_t;
typedef __attribute__((ext_vector_type(16))) _Float16 v16h;
typedef __attribute__((ext_vector_type(8)))  _Float16 v8h;
typedef __attribute__((ext_vector_type(8)))  float    v8f;

union AFrag {
    v16h v;
    v8h  h[2];
};

#define WPB 8  // waves per block (256 threads, wave32)

// ------------------------- elementwise helpers -----------------------------

__global__ void fill_h0_kernel(half_t* __restrict__ p, long n) {
    long i = (long)blockIdx.x * 256 + threadIdx.x;
    if (i < n) p[i] = (half_t)0.0f;
}

__global__ void fill_f0_kernel(float* __restrict__ p, long n) {
    long i = (long)blockIdx.x * 256 + threadIdx.x;
    if (i < n) p[i] = 0.0f;
}

// Scatter points into the dense 64^3 x 32(padded from 3) f16 grid + mask.
__global__ void scatter_kernel(const float* __restrict__ feats,
                               const int* __restrict__ coors,
                               half_t* __restrict__ dense,
                               float* __restrict__ mask, int n) {
    int i = blockIdx.x * 256 + threadIdx.x;
    if (i >= n) return;
    int z = coors[4 * i + 1];
    int y = coors[4 * i + 2];
    int x = coors[4 * i + 3];
    long v = ((long)z * 64 + y) * 64 + x;
    mask[v] = 1.0f;
    dense[v * 32 + 0] = (half_t)feats[3 * i + 0];
    dense[v * 32 + 1] = (half_t)feats[3 * i + 1];
    dense[v * 32 + 2] = (half_t)feats[3 * i + 2];
}

// Swizzle f32 weights (3,5,5,Cin,Cout) into f16 WMMA-B fragments:
//   wf[((t*nk + kci)*Ntiles + nci)*512 + lane*16 + e]
//     = W[t][kci*32 + (e&7) + 16*(e>=8) + 8*(lane>=16)][nci*16 + (lane&15)]
__global__ void wswizzle_kernel(const float* __restrict__ w,
                                half_t* __restrict__ wf,
                                int Cin, int CinPad, int Cout) {
    long idx = (long)blockIdx.x * 256 + threadIdx.x;
    long total = 75L * CinPad * Cout;
    if (idx >= total) return;
    int  e    = (int)(idx & 15);
    int  lane = (int)((idx >> 4) & 31);
    long q    = idx >> 9;                 // fragment tile index
    int  Ntl  = Cout >> 4;
    int  nci  = (int)(q % Ntl);  q /= Ntl;
    int  nk   = CinPad >> 5;
    int  kci  = (int)(q % nk);
    int  t    = (int)(q / nk);
    int  k    = (e & 7) + ((e >= 8) ? 16 : 0) + ((lane >= 16) ? 8 : 0);
    int  cin  = kci * 32 + k;
    int  cout = nci * 16 + (lane & 15);
    float val = (cin < Cin) ? w[((long)t * Cin + cin) * Cout + cout] : 0.0f;
    wf[idx] = (half_t)val;
}

// Masked 2x2x2 maxpool (stride 2) + mask pool.
__global__ void pool2_kernel(const half_t* __restrict__ in,
                             const float* __restrict__ mIn,
                             half_t* __restrict__ out,
                             float* __restrict__ mOut,
                             int Sout, int C) {
    long idx = (long)blockIdx.x * 256 + threadIdx.x;
    long total = (long)Sout * Sout * Sout * C;
    if (idx >= total) return;
    int  c  = (int)(idx % C);
    long vo = idx / C;
    int  Sin = Sout * 2;
    int  zo = (int)(vo / ((long)Sout * Sout));
    int  rr = (int)(vo % ((long)Sout * Sout));
    int  yo = rr / Sout, xo = rr % Sout;
    float best = -1e30f, m2 = 0.0f;
    for (int dz = 0; dz < 2; ++dz)
        for (int dy = 0; dy < 2; ++dy)
            for (int dx = 0; dx < 2; ++dx) {
                long vi = ((long)(2 * zo + dz) * Sin + (2 * yo + dy)) * Sin + (2 * xo + dx);
                float mk = mIn[vi];
                float vl = (mk > 0.0f) ? (float)in[vi * C + c] : -1e30f;
                best = fmaxf(best, vl);
                m2   = fmaxf(m2, mk);
            }
    out[vo * C + c] = (half_t)((m2 > 0.0f) ? best : 0.0f);
    if (c == 0) mOut[vo] = m2;
}

// ------------------------- implicit-GEMM conv ------------------------------
// 3x5x5 conv, pad (1,2,2), NDHWC f16, masked output.
// Wave: 16 voxels x 32 couts. Block: 8 M-tiles x one 32-cout N-group.
__global__ void conv3d_wmma_kernel(const half_t* __restrict__ in,
                                   half_t* __restrict__ out,
                                   float* __restrict__ fout,
                                   const float* __restrict__ mask,
                                   const half_t* __restrict__ wfrag,
                                   const half_t* __restrict__ zpad,
                                   int S, int CinPad, int Cout,
                                   int Mtiles, int NG) {
    __shared__ half_t bstage[8192];            // <=8 K-chunks x 2 tiles x 512 h
    __shared__ float  otile[WPB][16][32];

    int lane = threadIdx.x & 31;
    int wv   = threadIdx.x >> 5;
    int ng   = blockIdx.x % NG;                // 32-cout group
    int mt   = (blockIdx.x / NG) * WPB + wv;
    bool mvalid = mt < Mtiles;
    int  mtc = mvalid ? mt : (Mtiles - 1);     // clamp: keep all waves in step
    int  Ntiles = NG * 2;
    int  nt0 = ng * 2;

    int m     = lane & 15;
    int voxel = mtc * 16 + m;
    int S2 = S * S, S3 = S2 * S;
    bool vvalid = voxel < S3;
    int z = 0, y = 0, x = 0;
    if (vvalid) {
        z = voxel / S2;
        int r = voxel - z * S2;
        y = r / S;
        x = r - y * S;
    }
    int dlt = (lane >= 16) ? 8 : 0;   // f16 A layout: hi half-wave carries K+8
    int nk  = CinPad >> 5;
    int chunks = nk * 128;            // 16B chunks staged per tap

    v8f acc0 = {}, acc1 = {};

    for (int t = 0; t < 75; ++t) {
        int dz = t / 25, rem = t - dz * 25;
        int dy = rem / 5, dx = rem - dy * 5;
        int iz = z + dz - 1, iy = y + dy - 2, ix = x + dx - 2;
        bool inb = vvalid && (unsigned)iz < (unsigned)S &&
                   (unsigned)iy < (unsigned)S && (unsigned)ix < (unsigned)S;
        long base = ((long)(iz * S + iy) * S + ix) * CinPad;
        // Zero-page clamp: OOB lanes read from a 512B zeroed region instead of
        // branching -> unconditional loads, full EXEC at every WMMA.
        const half_t* apBase = inb ? (in + base + dlt) : (zpad + dlt);

        // ---- stage this tap's weight fragments into LDS (async DMA) ----
        __syncthreads();   // previous tap's LDS reads complete
        const half_t* wsrc = wfrag + ((long)(t * nk) * Ntiles + nt0) * 512;
        for (int c = threadIdx.x; c < chunks; c += 256) {
            int kci = c >> 7;
            int wi8 = (c & 127) * 8;
            const half_t* src = wsrc + (long)kci * Ntiles * 512 + wi8;
            unsigned lofs = (unsigned)(unsigned long long)&bstage[kci * 1024 + wi8];
            unsigned long long ga = (unsigned long long)src;
            asm volatile("global_load_async_to_lds_b128 %0, %1, off"
                         :: "v"(lofs), "v"(ga) : "memory");
        }
        if (t < 74 && threadIdx.x == 0)        // warm next tap's weights
            __builtin_prefetch((const void*)(wsrc + (long)nk * Ntiles * 512), 0, 1);
        asm volatile("s_wait_asynccnt 0x0" ::: "memory");
        __syncthreads();

        // ---- compute: nk chunks x 2 WMMAs, A shared between both ----
        for (int kci = 0; kci < nk; ++kci) {
            AFrag af;
            const half_t* ap = apBase + (kci << 5);
            af.h[0] = *reinterpret_cast<const v8h*>(ap);
            af.h[1] = *reinterpret_cast<const v8h*>(ap + 16);
            v16h b0 = *reinterpret_cast<const v16h*>(&bstage[kci * 1024 + lane * 16]);
            v16h b1 = *reinterpret_cast<const v16h*>(&bstage[kci * 1024 + 512 + lane * 16]);
            acc0 = __builtin_amdgcn_wmma_f32_16x16x32_f16(
                false, af.v, false, b0, (short)0, acc0, false, false);
            acc1 = __builtin_amdgcn_wmma_f32_16x16x32_f16(
                false, af.v, false, b1, (short)0, acc1, false, false);
        }
    }

    // ---- epilogue: stage 16x32 f32 tile in LDS, packed masked stores ----
    int rbase = (lane >= 16) ? 8 : 0;
    for (int v = 0; v < 8; ++v) {
        otile[wv][rbase + v][m]      = acc0[v];
        otile[wv][rbase + v][m + 16] = acc1[v];
    }
    // Same-wave DS ordering; per-wave private region -> no barrier needed.

    int r  = lane >> 1;
    int cb = (lane & 1) * 16;
    int vox = mtc * 16 + r;
    if (mvalid && vox < S3) {
        float mv = mask[vox];
        long  obase = (long)vox * Cout + nt0 * 16 + cb;
        if (fout) {
            for (int j = 0; j < 16; ++j)
                fout[obase + j] = otile[wv][r][cb + j] * mv;
        } else {
            v8h pk0, pk1;
            for (int j = 0; j < 8; ++j) {
                pk0[j] = (half_t)(otile[wv][r][cb + j] * mv);
                pk1[j] = (half_t)(otile[wv][r][cb + 8 + j] * mv);
            }
            *reinterpret_cast<v8h*>(out + obase)     = pk0;
            *reinterpret_cast<v8h*>(out + obase + 8) = pk1;
        }
    }
}

// ------------------------------- host side ---------------------------------

static inline int cdiv(long a, long b) { return (int)((a + b - 1) / b); }

extern "C" void kernel_launch(void* const* d_in, const int* in_sizes, int n_in,
                              void* d_out, int out_size, void* d_ws, size_t ws_size,
                              hipStream_t stream) {
    (void)n_in; (void)out_size; (void)ws_size;

    const float* feats = (const float*)d_in[0];
    const int*   coors = (const int*)d_in[1];
    // d_in[2] = batch_size (always 1)
    const float* W[14];
    for (int i = 0; i < 14; ++i) W[i] = (const float*)d_in[3 + i];
    int npts = in_sizes[1] / 4;

    // Workspace (~80 MB): act0/act1 (33.5 MB each), mask0/mask1 (1 MB each),
    // zpad (1 KB zero page), wfrag (9.8 MB, largest layer 75x256x256 f16).
    char* ws = (char*)d_ws;
    const size_t ACT_B = 33554432;
    half_t* act0  = (half_t*)(ws);
    half_t* act1  = (half_t*)(ws + ACT_B);
    float*  mask0 = (float*)(ws + 2 * ACT_B);
    float*  mask1 = (float*)(ws + 2 * ACT_B + 1048576);
    half_t* zpad  = (half_t*)(ws + 2 * ACT_B + 2097152);
    half_t* wfrag = (half_t*)(ws + 2 * ACT_B + 2097152 + 1024);

    long n0 = 262144L * 32;
    fill_h0_kernel<<<cdiv(n0, 256), 256, 0, stream>>>(act0, n0);
    fill_f0_kernel<<<cdiv(262144, 256), 256, 0, stream>>>(mask0, 262144);
    fill_h0_kernel<<<1, 256, 0, stream>>>(zpad, 512);
    scatter_kernel<<<cdiv(npts, 256), 256, 0, stream>>>(feats, coors, act0, mask0, npts);

    static const int convCin[14]  = {3, 64, 64, 96, 96, 128, 128, 160, 160, 192, 192, 224, 224, 256};
    static const int convCinP[14] = {32, 64, 64, 96, 96, 128, 128, 160, 160, 192, 192, 224, 224, 256};
    static const int convCout[14] = {64, 64, 96, 96, 128, 128, 160, 160, 192, 192, 224, 224, 256, 256};

    half_t* aIn = act0;  half_t* aOut = act1;
    float*  mIn = mask0; float*  mOut = mask1;
    int S = 64;

    auto runConv = [&](int wi, bool toFloat) {
        long wtot = 75L * convCinP[wi] * convCout[wi];
        wswizzle_kernel<<<cdiv(wtot, 256), 256, 0, stream>>>(
            W[wi], wfrag, convCin[wi], convCinP[wi], convCout[wi]);
        int S3 = S * S * S;
        int Mt = (S3 + 15) / 16;
        int NG = convCout[wi] / 32;            // 32-cout groups
        int blocks = cdiv(Mt, WPB) * NG;
        conv3d_wmma_kernel<<<blocks, 256, 0, stream>>>(
            aIn,
            toFloat ? nullptr : aOut,
            toFloat ? (float*)d_out : nullptr,
            mIn, wfrag, zpad, S, convCinP[wi], convCout[wi], Mt, NG);
        if (!toFloat) { half_t* t = aIn; aIn = aOut; aOut = t; }
    };

    int wi = 0;
    for (int lvl = 0; lvl < 6; ++lvl) {
        runConv(wi++, false);
        runConv(wi++, false);
        int So = S / 2;
        int C  = convCout[wi - 1];
        long ptot = (long)So * So * So * C;
        pool2_kernel<<<cdiv(ptot, 256), 256, 0, stream>>>(aIn, mIn, aOut, mOut, So, C);
        { half_t* t = aIn; aIn = aOut; aOut = t; }
        { float* tm = mIn; mIn = mOut; mOut = tm; }
        S = So;
    }
    // At S=1 the (3,5,5)/pad(1,2,2) mask dilation is the identity -> reuse mIn.
    runConv(12, false);
    runConv(13, true);   // writes f32 directly to d_out (256 values)
}
